// GNN_4080218931959
// MI455X (gfx1250) — compile-verified, matrix operand order checked by reading, not run on hardware
//
#include <hip/hip_runtime.h>

#define B_   128
#define NO_  256
#define NR_  512
#define D_   1024

typedef _Float16 v16h __attribute__((ext_vector_type(16)));
typedef _Float16 h8   __attribute__((ext_vector_type(8)));
typedef float    v8f  __attribute__((ext_vector_type(8)));
typedef int      i4   __attribute__((vector_size(16)));   // gcc-style, matches builtin proto

#if __has_builtin(__builtin_amdgcn_global_load_async_to_lds_b128) && \
    __has_builtin(__builtin_amdgcn_s_wait_asynccnt)
#define USE_ASYNC_B 1
#else
#define USE_ASYNC_B 0
#endif

// ---------------- weight transpose + f16 convert: W[K][N] f32 -> Wt[N][K] f16
__global__ __launch_bounds__(256)
void wt_transpose(const float* __restrict__ W, _Float16* __restrict__ Wt,
                  int K, int N) {
  __shared__ float t[32][33];
  const int n0 = blockIdx.x * 32;
  const int k0 = blockIdx.y * 32;
  const int tx = threadIdx.x & 31;
  const int tg = threadIdx.x >> 5;           // 0..7
  for (int i = 0; i < 4; ++i) {
    int kk = tg * 4 + i;                     // 0..31
    t[kk][tx] = W[(size_t)(k0 + kk) * N + n0 + tx];
  }
  __syncthreads();
  for (int i = 0; i < 4; ++i) {
    int nn = tg * 4 + i;
    Wt[(size_t)(n0 + nn) * K + k0 + tx] = (_Float16)t[tx][nn];
  }
}

// ---------------- new_obj = obj (straight copy)
__global__ __launch_bounds__(256)
void copy_f4(const float4* __restrict__ src, float4* __restrict__ dst, int n4) {
  int i = blockIdx.x * 256 + threadIdx.x;
  if (i < n4) dst[i] = src[i];
}

// ---------------- segmented-K WMMA GEMM, double-buffered LDS pipeline
// A row = concat of 1024-wide segments:
//   !RELA: [obj2[row], attr2[row]]                      (KTOT = 2048)
//    RELA: [obj2[eS[row]], rela2[row], obj2[eO[row]]]   (KTOT = 3072)
// out[row][n] = relu(A@W + bias)[row][n] + resid[row][n]  (* mask[row] if RELA)
template<int KTOT, bool RELA>
__global__ __launch_bounds__(256)
void gemm_seg(const float* __restrict__ seg0, const float* __restrict__ seg1,
              const long long* __restrict__ edges,
              const _Float16* __restrict__ Wt, const float* __restrict__ bias,
              const float* __restrict__ resid, const float* __restrict__ mask,
              float* __restrict__ out) {
  __shared__ _Float16 As[2][128 * 40];   // [row][k] padded to 40 halfs (80B rows)
  __shared__ _Float16 Bs[2][128 * 40];   // [col][k] padded
  __shared__ int idxS[128], idxO[128];

  const int gr0  = blockIdx.y * 128;  // global row base
  const int n0   = blockIdx.x * 128;  // global col base
  const int tid  = threadIdx.x;
  const int lane = tid & 31;
  const int wid  = tid >> 5;          // 8 waves
  const int wm   = wid >> 1;          // 0..3  -> 32-row strip
  const int wn   = wid & 1;           // 0..1  -> 64-col strip

  if (RELA && tid < 128) {
    int gr = gr0 + tid;
    int b  = gr >> 9;                             // gr / NR_
    idxS[tid] = b * NO_ + (int)edges[(size_t)gr * 2 + 0];
    idxO[tid] = b * NO_ + (int)edges[(size_t)gr * 2 + 1];
  }
  __syncthreads();   // idx visible before first stage (harmless for attr)

  v8f acc[2][4];
  for (int i = 0; i < 2; ++i)
    for (int j = 0; j < 4; ++j)
      acc[i][j] = (v8f){0.f, 0.f, 0.f, 0.f, 0.f, 0.f, 0.f, 0.f};

  const int ar  = tid >> 1;           // A staging: row 0..127
  const int akh = (tid & 1) * 16;     // k-half within 32

  auto stage = [&](int buf, int kk) {
    // ---- A tile: 128x32 fp32 -> f16 LDS (segmented / gathered rows)
    {
      const int seg = kk >> 10;
      const int kl  = (kk & 1023) + akh;
      const float* src;
      if (!RELA) {
        src = (seg == 0 ? seg0 : seg1) + (size_t)(gr0 + ar) * D_ + kl;
      } else if (seg == 1) {
        src = seg1 + (size_t)(gr0 + ar) * D_ + kl;
      } else {
        int ridx = (seg == 0) ? idxS[ar] : idxO[ar];
        src = seg0 + (size_t)ridx * D_ + kl;
      }
      const float4* s4 = (const float4*)src;
      float4 f0 = s4[0], f1 = s4[1], f2 = s4[2], f3 = s4[3];
      h8 v0 = {(_Float16)f0.x, (_Float16)f0.y, (_Float16)f0.z, (_Float16)f0.w,
               (_Float16)f1.x, (_Float16)f1.y, (_Float16)f1.z, (_Float16)f1.w};
      h8 v1 = {(_Float16)f2.x, (_Float16)f2.y, (_Float16)f2.z, (_Float16)f2.w,
               (_Float16)f3.x, (_Float16)f3.y, (_Float16)f3.z, (_Float16)f3.w};
      *(h8*)&As[buf][ar * 40 + akh]     = v0;
      *(h8*)&As[buf][ar * 40 + akh + 8] = v1;
    }
    // ---- B tile: 128 cols x 32 k, already f16 [N][K] in global
    {
      int c = tid * 2;
      for (int q = 0; q < 2; ++q, ++c) {
        int n  = c >> 2;
        int kc = (c & 3) * 8;
        const _Float16* g = Wt + (size_t)(n0 + n) * KTOT + kk + kc;
        _Float16*       l = &Bs[buf][n * 40 + kc];
#if USE_ASYNC_B
        __builtin_amdgcn_global_load_async_to_lds_b128(
            (__attribute__((address_space(1))) i4*)g,
            (__attribute__((address_space(3))) i4*)l, 0, 0);
#else
        *(h8*)l = *(const h8*)g;
#endif
      }
    }
  };

  stage(0, 0);   // prologue fill

  const int lr  = lane & 15;
  const int kof = (lane < 16) ? 0 : 8;

  for (int k0 = 0; k0 < KTOT; k0 += 32) {
    const int buf = (k0 >> 5) & 1;
#if USE_ASYNC_B
    __builtin_amdgcn_s_wait_asynccnt(0);   // async B copies landed in LDS
#endif
    __syncthreads();                       // publish staged buffer

    if (k0 + 32 < KTOT) stage(buf ^ 1, k0 + 32);   // overlap next tile with WMMA

    // ---- fragment loads per ISA 16-bit A/B layout + 8 WMMAs
    v16h af[2], bf[4];
    for (int i = 0; i < 2; ++i) {
      int row = wm * 32 + i * 16 + lr;
      ((h8*)&af[i])[0] = *(h8*)&As[buf][row * 40 + kof];
      ((h8*)&af[i])[1] = *(h8*)&As[buf][row * 40 + kof + 16];
    }
    for (int j = 0; j < 4; ++j) {
      int col = wn * 64 + j * 16 + lr;
      ((h8*)&bf[j])[0] = *(h8*)&Bs[buf][col * 40 + kof];
      ((h8*)&bf[j])[1] = *(h8*)&Bs[buf][col * 40 + kof + 16];
    }
    for (int i = 0; i < 2; ++i)
      for (int j = 0; j < 4; ++j)
        acc[i][j] = __builtin_amdgcn_wmma_f32_16x16x32_f16(
            false, af[i], false, bf[j], (short)0, acc[i][j], false, false);
  }

  // ---- epilogue: bias + relu + residual (+ mask), fp32 store
  const int mhi = (lane >> 4) * 8;   // C/D layout: lanes 16-31 hold M=8+r
  for (int i = 0; i < 2; ++i)
    for (int j = 0; j < 4; ++j) {
      int   gn = n0 + wn * 64 + j * 16 + lr;
      float bv = bias[gn];
      for (int e = 0; e < 8; ++e) {
        int   gm = gr0 + wm * 32 + i * 16 + mhi + e;
        float v  = acc[i][j][e] + bv;
        v = v > 0.f ? v : 0.f;
        v += resid[(size_t)gm * D_ + gn];
        if (RELA) v *= mask[gm];
        out[(size_t)gm * D_ + gn] = v;
      }
    }
}

extern "C" void kernel_launch(void* const* d_in, const int* in_sizes, int n_in,
                              void* d_out, int out_size, void* d_ws, size_t ws_size,
                              hipStream_t stream) {
  const float*     obj   = (const float*)d_in[0];
  const float*     attr  = (const float*)d_in[1];
  const float*     rela  = (const float*)d_in[2];
  const long long* edges = (const long long*)d_in[3];   // int64 in reference
  const float*     mask  = (const float*)d_in[4];
  const float*     Wa    = (const float*)d_in[5];
  const float*     ba    = (const float*)d_in[6];
  const float*     Wr    = (const float*)d_in[7];
  const float*     br    = (const float*)d_in[8];
  float*           out   = (float*)d_out;

  // f16 transposed weights in workspace: Wt_attr (4MB) + Wt_rela (6MB)
  _Float16* wtA = (_Float16*)d_ws;
  _Float16* wtR = wtA + (size_t)2048 * 1024;

  wt_transpose<<<dim3(1024 / 32, 2048 / 32), 256, 0, stream>>>(Wa, wtA, 2048, 1024);
  wt_transpose<<<dim3(1024 / 32, 3072 / 32), 256, 0, stream>>>(Wr, wtR, 3072, 1024);

  // new_obj = obj
  const int n4 = B_ * NO_ * D_ / 4;
  copy_f4<<<(n4 + 255) / 256, 256, 0, stream>>>((const float4*)obj, (float4*)out, n4);

  float* outAttr = out + (size_t)B_ * NO_ * D_;
  float* outRela = outAttr + (size_t)B_ * NO_ * D_;

  // attr MLP: M=32768, K=2048, N=1024
  gemm_seg<2048, false><<<dim3(1024 / 128, (B_ * NO_) / 128), 256, 0, stream>>>(
      obj, attr, nullptr, wtA, ba, attr, nullptr, outAttr);

  // rela MLP: M=65536, K=3072, N=1024 (gathered s/o segments)
  gemm_seg<3072, true><<<dim3(1024 / 128, (B_ * NR_) / 128), 256, 0, stream>>>(
      obj, rela, edges, wtR, br, rela, mask, outRela);
}
